// PWCDCNet_15393162789088
// MI455X (gfx1250) — compile-verified, matrix operand order checked by reading, not run on hardware
//
#include <hip/hip_runtime.h>
#include <hip/hip_bf16.h>
#include <cstdint>
#include <cstddef>

// ---------------------------------------------------------------------------
// PWC-Net forward for MI455X (gfx1250, wave32).
// Conv layers = implicit-GEMM on V_WMMA_F32_16X16X32_F16 (fp16 in, fp32 acc),
// weight tiles double-buffered into LDS by the Tensor Data Mover (pipelined,
// s_wait_tensorcnt 1) when available. Correlation / warp / concat =
// bandwidth-bound elementwise kernels (23.3 TB/s HBM roofline).
// ---------------------------------------------------------------------------

typedef _Float16 f16;
typedef __attribute__((ext_vector_type(16))) _Float16 v16h;
typedef __attribute__((ext_vector_type(8)))  _Float16 v8h;
typedef __attribute__((ext_vector_type(8)))  float    v8f;
typedef __attribute__((ext_vector_type(4)))  unsigned int u32x4;
typedef __attribute__((ext_vector_type(8)))  int i32x8;
typedef __attribute__((ext_vector_type(4)))  int i32x4;

#if __has_builtin(__builtin_amdgcn_tensor_load_to_lds) && __has_builtin(__builtin_amdgcn_s_wait_tensorcnt)
#define USE_TDM 1
#else
#define USE_TDM 0
#endif

// ---------------- small utility kernels ----------------

__global__ void k_f32_to_f16(const float* __restrict__ s, f16* __restrict__ d, int n) {
    int i = blockIdx.x * blockDim.x + threadIdx.x;
    if (i < n) d[i] = (f16)s[i];
}

__global__ void k_slice_f16(const float* __restrict__ src, f16* __restrict__ dst,
                            int B, int coff, int Ctot, int C, int P) {
    int i = blockIdx.x * blockDim.x + threadIdx.x;
    int tot = B * C * P;
    if (i >= tot) return;
    int m = i % P;
    int c = (i / P) % C;
    int b = i / (P * C);
    dst[i] = (f16)src[((size_t)b * Ctot + coff + c) * P + m];
}

__global__ void k_copy_ch(const f16* __restrict__ src, f16* __restrict__ dst,
                          int B, int Cs, int P, int Ct, int off) {
    int i = blockIdx.x * blockDim.x + threadIdx.x;
    int tot = B * Cs * P;
    if (i >= tot) return;
    int m = i % P;
    int c = (i / P) % Cs;
    int b = i / (P * Cs);
    dst[((size_t)b * Ct + off + c) * P + m] = src[i];
}

__global__ void k_add_out(const f16* __restrict__ a, const f16* __restrict__ b,
                          float* __restrict__ o, int n) {
    int i = blockIdx.x * blockDim.x + threadIdx.x;
    if (i < n) o[i] = (float)a[i] + (float)b[i];
}

// ---------------- correlation (cost volume), fused leaky-ReLU ----------------
__global__ void k_corr(const f16* __restrict__ f1, const f16* __restrict__ f2,
                       f16* __restrict__ out, int B, int C, int H, int W) {
    int P = H * W;
    int i = blockIdx.x * blockDim.x + threadIdx.x;
    int tot = B * 81 * P;
    if (i >= tot) return;
    int m = i % P;
    int d = (i / P) % 81;
    int b = i / (P * 81);
    int x = m % W, y = m / W;
    int dy = d / 9 - 4, dx = d % 9 - 4;
    int y2 = y + dy, x2 = x + dx;
    float s = 0.f;
    if (y2 >= 0 && y2 < H && x2 >= 0 && x2 < W) {
        const f16* A = f1 + (size_t)b * C * P + (size_t)y * W + x;
        const f16* Bp = f2 + (size_t)b * C * P + (size_t)y2 * W + x2;
        #pragma unroll 4
        for (int c = 0; c < C; c++)
            s += (float)A[(size_t)c * P] * (float)Bp[(size_t)c * P];
    }
    s *= (1.f / (float)C);
    s = s > 0.f ? s : 0.1f * s;
    out[((size_t)b * 81 + d) * P + m] = (f16)s;
}

// ---------------- bilinear warp with zero padding + occlusion mask ----------------
__global__ void k_warp(const f16* __restrict__ feat, const f16* __restrict__ flow,
                       f16* __restrict__ out, int B, int C, int H, int W, float scale) {
    int P = H * W;
    int i = blockIdx.x * blockDim.x + threadIdx.x;
    int tot = B * C * P;
    if (i >= tot) return;
    int m = i % P;
    int c = (i / P) % C;
    int b = i / (P * C);
    int x = m % W, y = m / W;
    float fx = (float)flow[((size_t)b * 2 + 0) * P + m] * scale;
    float fy = (float)flow[((size_t)b * 2 + 1) * P + m] * scale;
    float px = (float)x + fx;
    float py = (float)y + fy;
    float x0 = floorf(px), y0 = floorf(py);
    float wx = px - x0, wy = py - y0;
    const f16* F = feat + ((size_t)b * C + c) * P;
    float acc = 0.f, msk = 0.f;
    #pragma unroll
    for (int k = 0; k < 4; k++) {
        float cx = x0 + (float)(k & 1);
        float cy = y0 + (float)(k >> 1);
        float w = ((k & 1) ? wx : 1.f - wx) * ((k >> 1) ? wy : 1.f - wy);
        bool valid = (cx >= 0.f) && (cx <= (float)(W - 1)) && (cy >= 0.f) && (cy <= (float)(H - 1));
        if (valid) {
            int ix = (int)cx, iy = (int)cy;
            acc += (float)F[(size_t)iy * W + ix] * w;
            msk += w;
        }
    }
    out[i] = (f16)(msk >= 0.9999f ? acc : 0.f);
}

#if USE_TDM
// Issue a TDM 2D tile load: 32(K) x 64(N) f16 from a row-major [Cout][K]
// weight tensor into LDS at lds_addr, with hardware padding of 4 DWORDs
// every 16 DWORDs (-> 40-halfword LDS pitch). D# per CDNA5 ISA ch.8.
__device__ __forceinline__ void tdm_issue_weights(const f16* Wt, int n0, int K, int k0,
                                                  int Cout, unsigned lds_addr) {
    unsigned long long gaddr = (unsigned long long)(uintptr_t)(Wt + (size_t)n0 * K + k0);
    int remK = K - k0;
    int remN = Cout - n0;
    u32x4 g0 = {};
    g0[0] = 1u;                                    // count=1 (valid descriptor)
    g0[1] = lds_addr;                              // lds_addr
    g0[2] = (unsigned)gaddr;                       // global_addr low
    g0[3] = (unsigned)(gaddr >> 32) | (2u << 30);  // global_addr hi | type=2
    i32x8 g1 = {};
    g1[0] = (1 << 16) | (1 << 20) | (3 << 22) | (3 << 25); // 2B elems, pad en, ivl=16dw, amt=4dw
    g1[1] = (remK & 0xFFFF) << 16;                          // tensor_dim0 lo16
    g1[2] = ((remK >> 16) & 0xFFFF) | ((remN & 0xFFFF) << 16);
    g1[3] = ((remN >> 16) & 0xFFFF) | (32 << 16);           // tile_dim0=32
    g1[4] = 64;                                             // tile_dim1=64, tile_dim2=0
    g1[5] = K;                                              // tensor_dim0_stride lo32
    g1[6] = 0;
    g1[7] = 0;
    i32x4 z4 = {};
#if defined(__clang_major__) && (__clang_major__ >= 23)
    i32x8 z8 = {};
    __builtin_amdgcn_tensor_load_to_lds(g0, g1, z4, z4, z8, 0);
#else
    __builtin_amdgcn_tensor_load_to_lds(g0, g1, z4, z4, 0);
#endif
}
#endif

// ---------------- implicit-GEMM conv on WMMA ----------------
// GEMM: M = OH*OW pixels, N = Cout, K = Cin*KH*KW.
// Tile BM=128, BN=64, BK=32; 256 threads = 8 wave32s; each wave owns a
// 32(M) x 32(N) slab -> 4 x v_wmma_f32_16x16x32_f16 per K-slab with A/B
// fragment reuse. Weight tiles are double-buffered via TDM: keep up to 2
// descriptors in flight, s_wait_tensorcnt(1) hides DMA latency under WMMA.
__global__ __launch_bounds__(256)
void k_conv_wmma(const f16* __restrict__ X, const f16* __restrict__ Wt,
                 const float* __restrict__ bias, f16* __restrict__ Y,
                 int Cin, int H, int W, int Cout, int OH, int OW,
                 int KH, int KW, int strd, int pad, int dil,
                 int transposed, int act, int outCtot, int outCoff) {
    __shared__ __align__(16) f16 As[128][40];     // [M][K], pitch 40 = 16B-aligned
    __shared__ __align__(16) f16 Bs[2][64][40];   // ping-pong [N][K] weight tiles
    __shared__ int s_ioy[16];
    __shared__ int s_iox[16];

    const int b  = blockIdx.z;
    const int m0 = blockIdx.x * 128;
    const int n0 = blockIdx.y * 64;
    const int P   = OH * OW;
    const int KHW = KH * KW;
    const int K   = Cin * KHW;

    const int tid  = threadIdx.x;
    const int lane = tid & 31;
    const int wv   = tid >> 5;
    const int mt   = wv & 3;   // 32-row M slab
    const int nt   = wv >> 2;  // 32-col N slab

    // per-tap offset tables (<=16 entries), computed once
    if (tid < KHW) {
        int ky = tid / KW, kx = tid - (tid / KW) * KW;
        if (!transposed) { s_ioy[tid] = ky * dil - pad; s_iox[tid] = kx * dil - pad; }
        else             { s_ioy[tid] = ky - pad;       s_iox[tid] = kx - pad; }
    }

    const f16* Xb = X + (size_t)b * Cin * H * W;

    v8f acc00 = {}, acc01 = {}, acc10 = {}, acc11 = {};

    // A cooperative-load mapping: row = tid&127, 16 consecutive K elements
    const int ar = tid & 127;
    const int ak = (tid >> 7) * 16;
    const int mm = m0 + ar;
    const bool mok = mm < P;
    int oy = 0, ox = 0;
    if (mok) { oy = mm / OW; ox = mm - oy * OW; }
    const int oys = transposed ? oy : oy * strd;
    const int oxs = transposed ? ox : ox * strd;
    int cc = ak / KHW;            // carry-propagated k -> (channel, tap)
    int rr = ak - cc * KHW;

#if !USE_TDM
    // B cooperative-load mapping (manual path): row = tid&63, 8 K elements
    const int br = tid & 63;
    const int bk = ((tid >> 6) & 3) * 8;
    const int co = n0 + br;
    const bool cok = co < Cout;
#else
    const unsigned ldsb = (unsigned)(unsigned long long)(uintptr_t)&Bs[0][0][0];
    const unsigned bufBytes = 64u * 40u * 2u;   // 5120 B per ping-pong buffer
    // prologue: first weight tile in flight before the loop
    if (tid == 0) tdm_issue_weights(Wt, n0, K, 0, Cout, ldsb);
#endif

    __syncthreads();  // s_ioy/s_iox visible

    int idx = 0;
    for (int k0 = 0; k0 < K; k0 += 32, idx ^= 1) {
        // ---- A tile: im2col gather, carry-propagated decomposition ----
        {
            v8h av0 = {}, av1 = {};
            int c2 = cc, r2 = rr;
            #pragma unroll
            for (int i = 0; i < 16; i++) {
                f16 v = (f16)0.f;
                int k = k0 + ak + i;
                if (mok && k < K) {
                    int yy = oys + s_ioy[r2];
                    int xx = oxs + s_iox[r2];
                    bool ok;
                    int iy, ix;
                    if (!transposed) {
                        iy = yy; ix = xx;
                        ok = ((unsigned)iy < (unsigned)H) && ((unsigned)ix < (unsigned)W);
                    } else {
                        ok = (yy >= 0) && (xx >= 0) && !(yy & 1) && !(xx & 1);
                        iy = yy >> 1; ix = xx >> 1;
                        ok = ok && (iy < H) && (ix < W);
                    }
                    if (ok) v = Xb[(size_t)c2 * H * W + (size_t)iy * W + ix];
                }
                if (i < 8) av0[i] = v; else av1[i - 8] = v;
                r2++; if (r2 == KHW) { r2 = 0; c2++; }
            }
            *(v8h*)&As[ar][ak]     = av0;
            *(v8h*)&As[ar][ak + 8] = av1;
        }
        rr += 32;
        while (rr >= KHW) { rr -= KHW; cc++; }

#if USE_TDM
        // ---- pipeline: issue next weight tile, wait only for the current one ----
        if (tid == 0 && (k0 + 32) < K)
            tdm_issue_weights(Wt, n0, K, k0 + 32, Cout,
                              ldsb + (unsigned)(idx ^ 1) * bufBytes);
        if (wv == 0) {
            if ((k0 + 32) < K) __builtin_amdgcn_s_wait_tensorcnt(1);
            else               __builtin_amdgcn_s_wait_tensorcnt(0);
        }
#else
        // ---- B tile: manual vector loads ([Cout][K] row-major) ----
        if (cok && (k0 + bk + 8) <= K) {
            *(v8h*)&Bs[idx][br][bk] = *(const v8h*)&Wt[(size_t)co * K + k0 + bk];
            if (k0 + 32 + bk < K)
                __builtin_prefetch(&Wt[(size_t)co * K + k0 + 32 + bk], 0, 1);
        } else {
            #pragma unroll
            for (int i = 0; i < 8; i++) {
                int k = k0 + bk + i;
                Bs[idx][br][bk + i] = (cok && k < K) ? Wt[(size_t)co * K + k] : (f16)0.f;
            }
        }
#endif
        __syncthreads();

        // ---- fragments (ISA 16-bit layout: lanes 0-15 K{0..7,16..23},
        //      lanes 16-31 K{8..15,24..31}) + 4 WMMAs with reuse ----
        const int kb = (lane < 16) ? 0 : 8;
        const int mr0 = mt * 32 + (lane & 15);
        v8h alo = *(const v8h*)&As[mr0][kb];
        v8h ahi = *(const v8h*)&As[mr0][kb + 16];
        v16h a0 = __builtin_shufflevector(alo, ahi, 0,1,2,3,4,5,6,7,8,9,10,11,12,13,14,15);
        v8h a1lo = *(const v8h*)&As[mr0 + 16][kb];
        v8h a1hi = *(const v8h*)&As[mr0 + 16][kb + 16];
        v16h a1 = __builtin_shufflevector(a1lo, a1hi, 0,1,2,3,4,5,6,7,8,9,10,11,12,13,14,15);

        const int nc = nt * 32 + (lane & 15);
        v8h b0lo = *(const v8h*)&Bs[idx][nc][kb];
        v8h b0hi = *(const v8h*)&Bs[idx][nc][kb + 16];
        v16h bf0 = __builtin_shufflevector(b0lo, b0hi, 0,1,2,3,4,5,6,7,8,9,10,11,12,13,14,15);
        v8h b1lo = *(const v8h*)&Bs[idx][nc + 16][kb];
        v8h b1hi = *(const v8h*)&Bs[idx][nc + 16][kb + 16];
        v16h bf1 = __builtin_shufflevector(b1lo, b1hi, 0,1,2,3,4,5,6,7,8,9,10,11,12,13,14,15);

        acc00 = __builtin_amdgcn_wmma_f32_16x16x32_f16(false, a0, false, bf0, (short)0, acc00, false, false);
        acc01 = __builtin_amdgcn_wmma_f32_16x16x32_f16(false, a0, false, bf1, (short)0, acc01, false, false);
        acc10 = __builtin_amdgcn_wmma_f32_16x16x32_f16(false, a1, false, bf0, (short)0, acc10, false, false);
        acc11 = __builtin_amdgcn_wmma_f32_16x16x32_f16(false, a1, false, bf1, (short)0, acc11, false, false);
        __syncthreads();
    }

    // ---- epilogue: bias + leaky-ReLU, f16 store at channel offset ----
    // C/D layout: VGPR r, lane L -> M = r + (L<16 ? 0 : 8), N = L&15
    const int ncol0 = n0 + nt * 32 + (lane & 15);
    const int ncol1 = ncol0 + 16;
    const float bia0 = (ncol0 < Cout) ? bias[ncol0] : 0.f;
    const float bia1 = (ncol1 < Cout) ? bias[ncol1] : 0.f;
    const int rof = (lane >> 4) << 3;
    #pragma unroll
    for (int r = 0; r < 8; r++) {
        int mA = m0 + mt * 32 + rof + r;
        int mB = mA + 16;
        if (mA < P) {
            if (ncol0 < Cout) {
                float v = acc00[r] + bia0;
                if (act) v = v > 0.f ? v : 0.1f * v;
                Y[((size_t)b * outCtot + outCoff + ncol0) * P + mA] = (f16)v;
            }
            if (ncol1 < Cout) {
                float v = acc01[r] + bia1;
                if (act) v = v > 0.f ? v : 0.1f * v;
                Y[((size_t)b * outCtot + outCoff + ncol1) * P + mA] = (f16)v;
            }
        }
        if (mB < P) {
            if (ncol0 < Cout) {
                float v = acc10[r] + bia0;
                if (act) v = v > 0.f ? v : 0.1f * v;
                Y[((size_t)b * outCtot + outCoff + ncol0) * P + mB] = (f16)v;
            }
            if (ncol1 < Cout) {
                float v = acc11[r] + bia1;
                if (act) v = v > 0.f ? v : 0.1f * v;
                Y[((size_t)b * outCtot + outCoff + ncol1) * P + mB] = (f16)v;
            }
        }
    }
}

// ---------------------------------------------------------------------------
// Host orchestration
// ---------------------------------------------------------------------------

extern "C" void kernel_launch(void* const* d_in, const int* in_sizes, int n_in,
                              void* d_out, int out_size, void* d_ws, size_t ws_size,
                              hipStream_t stream) {
    (void)in_sizes; (void)n_in; (void)out_size; (void)ws_size;
    const int B = 2;

    char* base = (char*)d_ws;
    size_t off = 0;
    auto alloc = [&](size_t bytes) -> void* {
        void* p = base + off;
        off = (off + bytes + 255) & ~(size_t)255;
        return p;
    };

    // params consumed in pytree (sorted-key) order:
    // c1..c6, conv2..conv6, dc, deconv, pf2..pf6, upfeat3..upfeat6
    int cur = 1;
    struct PW { const f16* w; const float* b; int co, ci, kh, kw; };
    auto grabc = [&](int co, int ci, int k) -> PW {
        const float* w32 = (const float*)d_in[cur++];
        const float* b32 = (const float*)d_in[cur++];
        size_t n = (size_t)co * ci * k * k;
        f16* w16 = (f16*)alloc(n * sizeof(f16));
        k_f32_to_f16<<<(int)((n + 255) / 256), 256, 0, stream>>>(w32, w16, (int)n);
        PW p; p.w = w16; p.b = b32; p.co = co; p.ci = ci; p.kh = k; p.kw = k;
        return p;
    };

    const int chs[7] = {3, 16, 32, 64, 96, 128, 196};
    PW c[6][3];
    for (int i = 0; i < 6; i++) {
        c[i][0] = grabc(chs[i + 1], chs[i], 3);
        c[i][1] = grabc(chs[i + 1], chs[i + 1], 3);
        c[i][2] = grabc(chs[i + 1], chs[i + 1], 3);
    }
    const int dd[5] = {128, 128, 96, 64, 32};
    int od_of[7]; od_of[2] = 117; od_of[3] = 149; od_of[4] = 181; od_of[5] = 213; od_of[6] = 81;
    PW convb[7][5];
    for (int lvl = 2; lvl <= 6; lvl++) {
        int cin = od_of[lvl];
        for (int j = 0; j < 5; j++) { convb[lvl][j] = grabc(dd[j], cin, 3); cin += dd[j]; }
    }
    PW dc[7];
    {
        const int shp[7][2] = {{128,565},{128,128},{128,128},{96,128},{64,96},{32,64},{2,32}};
        for (int j = 0; j < 7; j++) dc[j] = grabc(shp[j][0], shp[j][1], 3);
    }
    PW deconvP = grabc(2, 2, 4);
    PW pf[7];
    for (int lvl = 2; lvl <= 6; lvl++) pf[lvl] = grabc(2, od_of[lvl] + 448, 3);
    PW upfeat[7];
    for (int lvl = 3; lvl <= 6; lvl++) upfeat[lvl] = grabc(2, od_of[lvl] + 448, 4);

    // ---- layer launchers ----
    auto convRaw = [&](const f16* Xp, int Cin, int H, int Wd, const PW& pw,
                       int strd, int pad, int dil, int transposed, int act,
                       f16* Yp, int OH, int OW, int outCtot, int outCoff) {
        int P = OH * OW;
        dim3 g((P + 127) / 128, (pw.co + 63) / 64, B);
        k_conv_wmma<<<g, 256, 0, stream>>>(Xp, pw.w, pw.b, Yp, Cin, H, Wd, pw.co,
                                           OH, OW, pw.kh, pw.kw, strd, pad, dil,
                                           transposed, act, outCtot, outCoff);
    };
    auto convN = [&](const f16* Xp, int Cin, int H, int Wd, const PW& pw,
                     int strd, int pad, int dil, int act, int& OH, int& OW) -> f16* {
        OH = (H + 2 * pad - (dil * (pw.kh - 1) + 1)) / strd + 1;
        OW = (Wd + 2 * pad - (dil * (pw.kw - 1) + 1)) / strd + 1;
        f16* Yp = (f16*)alloc((size_t)B * pw.co * OH * OW * sizeof(f16));
        convRaw(Xp, Cin, H, Wd, pw, strd, pad, dil, 0, act, Yp, OH, OW, pw.co, 0);
        return Yp;
    };
    auto deconvN = [&](const f16* Xp, int Cin, int H, int Wd, const PW& pw) -> f16* {
        int OH = 2 * H, OW = 2 * Wd;
        f16* Yp = (f16*)alloc((size_t)B * pw.co * OH * OW * sizeof(f16));
        convRaw(Xp, Cin, H, Wd, pw, 1, 2, 1, 1, 0, Yp, OH, OW, pw.co, 0);
        return Yp;
    };
    auto copyCh = [&](const f16* src, int Cs, f16* dst, int Ct, int offc, int P) {
        int tot = B * Cs * P;
        k_copy_ch<<<(tot + 255) / 256, 256, 0, stream>>>(src, dst, B, Cs, P, Ct, offc);
    };
    auto block1f = [&](const f16* xin, int Cin, int H, int Wd, PW* ps,
                       int& oH, int& oW) -> f16* {
        int h1, w1, h2, w2;
        f16* a = convN(xin, Cin, H, Wd, ps[0], 2, 1, 1, 1, h1, w1);
        a = convN(a, ps[0].co, h1, w1, ps[1], 1, 1, 1, 1, h2, w2);
        a = convN(a, ps[1].co, h2, w2, ps[2], 1, 1, 1, 1, oH, oW);
        return a;
    };
    auto block2f = [&](f16*& xx, int& C, int H, int Wd, PW* ps) {
        int P = H * Wd;
        for (int j = 0; j < 5; j++) {
            int Cn = ps[j].co + C;
            f16* nb = (f16*)alloc((size_t)B * Cn * P * sizeof(f16));
            convRaw(xx, C, H, Wd, ps[j], 1, 1, 1, 0, 1, nb, H, Wd, Cn, 0);
            copyCh(xx, C, nb, Cn, ps[j].co, P);
            xx = nb; C = Cn;
        }
    };
    auto corrf = [&](const f16* A, const f16* Bb, int C, int H, int Wd) -> f16* {
        int P = H * Wd;
        f16* o = (f16*)alloc((size_t)B * 81 * P * sizeof(f16));
        int tot = B * 81 * P;
        k_corr<<<(tot + 255) / 256, 256, 0, stream>>>(A, Bb, o, B, C, H, Wd);
        return o;
    };
    auto warpf = [&](const f16* feat, int C, int H, int Wd, const f16* flo, float sc) -> f16* {
        int P = H * Wd;
        f16* o = (f16*)alloc((size_t)B * C * P * sizeof(f16));
        int tot = B * C * P;
        k_warp<<<(tot + 255) / 256, 256, 0, stream>>>(feat, flo, o, B, C, H, Wd, sc);
        return o;
    };

    // ---- inputs ----
    const float* x32 = (const float*)d_in[0];
    const int H0 = 384, W0 = 512, P0 = H0 * W0;
    f16* im1 = (f16*)alloc((size_t)B * 3 * P0 * sizeof(f16));
    f16* im2 = (f16*)alloc((size_t)B * 3 * P0 * sizeof(f16));
    {
        int tot = B * 3 * P0;
        k_slice_f16<<<(tot + 255) / 256, 256, 0, stream>>>(x32, im1, B, 0, 6, 3, P0);
        k_slice_f16<<<(tot + 255) / 256, 256, 0, stream>>>(x32, im2, B, 3, 6, 3, P0);
    }

    // ---- feature pyramids ----
    f16 *f1[6], *f2[6];
    int LH[6], LW[6];
    {
        const f16* a = im1; int C = 3, H = H0, Wd = W0;
        for (int i = 0; i < 6; i++) {
            int oh, ow;
            f16* r_ = block1f(a, C, H, Wd, c[i], oh, ow);
            f1[i] = r_; LH[i] = oh; LW[i] = ow;
            a = r_; C = chs[i + 1]; H = oh; Wd = ow;
        }
    }
    {
        const f16* a = im2; int C = 3, H = H0, Wd = W0;
        for (int i = 0; i < 6; i++) {
            int oh, ow;
            f16* r_ = block1f(a, C, H, Wd, c[i], oh, ow);
            f2[i] = r_;
            a = r_; C = chs[i + 1]; H = oh; Wd = ow;
        }
    }

    // ---- level 6 ----
    f16* xx = corrf(f1[5], f2[5], chs[6], LH[5], LW[5]);
    int Cxx = 81;
    block2f(xx, Cxx, LH[5], LW[5], convb[6]);
    int th, tw;
    f16* flow = convN(xx, Cxx, LH[5], LW[5], pf[6], 1, 1, 1, 0, th, tw);
    f16* up_flow = deconvN(flow, 2, LH[5], LW[5], deconvP);
    f16* up_feat = deconvN(xx, Cxx, LH[5], LW[5], upfeat[6]);

    // ---- levels 5..2 ----
    const int lvls[4] = {5, 4, 3, 2};
    const float scl[4] = {0.625f, 1.25f, 2.5f, 5.0f};
    int curH = LH[5], curW = LW[5];
    for (int t = 0; t < 4; t++) {
        int lvl = lvls[t], li = lvl - 1;
        int H = LH[li], Wd = LW[li], P = H * Wd, C1 = chs[lvl];
        f16* wp = warpf(f2[li], C1, H, Wd, up_flow, scl[t]);
        f16* cr = corrf(f1[li], wp, C1, H, Wd);
        int Ct = 81 + C1 + 4;
        f16* cat = (f16*)alloc((size_t)B * Ct * P * sizeof(f16));
        copyCh(cr, 81, cat, Ct, 0, P);
        copyCh(f1[li], C1, cat, Ct, 81, P);
        copyCh(up_flow, 2, cat, Ct, 81 + C1, P);
        copyCh(up_feat, 2, cat, Ct, 83 + C1, P);
        xx = cat; Cxx = Ct;
        block2f(xx, Cxx, H, Wd, convb[lvl]);
        flow = convN(xx, Cxx, H, Wd, pf[lvl], 1, 1, 1, 0, th, tw);
        if (lvl > 2) {
            up_flow = deconvN(flow, 2, H, Wd, deconvP);
            up_feat = deconvN(xx, Cxx, H, Wd, upfeat[lvl]);
        }
        curH = H; curW = Wd;
    }

    // ---- context network (dilated convs) + residual flow ----
    {
        int H = curH, Wd = curW;
        f16* r_ = xx; int Cr = Cxx;
        const int dils[6] = {1, 2, 4, 8, 16, 1};
        for (int j = 0; j < 6; j++) {
            int oh, ow;
            r_ = convN(r_, Cr, H, Wd, dc[j], 1, dils[j], dils[j], 1, oh, ow);
            Cr = dc[j].co;
        }
        int oh, ow;
        f16* resid = convN(r_, Cr, H, Wd, dc[6], 1, 1, 1, 0, oh, ow);
        int n = B * 2 * H * Wd;
        k_add_out<<<(n + 255) / 256, 256, 0, stream>>>(flow, resid, (float*)d_out, n);
    }
}
